// SinkhornDistance_7799660609867
// MI455X (gfx1250) — matrix operand order repaired and checked by reading, not compiled, not run on hardware
//
#include <hip/hip_runtime.h>
#include <hip/hip_bf16.h>
#include <stdint.h>

// ---------------------------------------------------------------------------
// Sinkhorn distance on MI455X (gfx1250, wave32, WMMA).
//   Phase 0: split x,y into bf16 hi/lo arrays (each element converted once)
//   Phase 1: C = 1 - x @ y^T   (bf16x3 WMMA, f32 accumulate, pure-load loop)
//            + LDS-tiled transpose producing CT (both L2-resident: 128 MiB)
//   Phase 2: 50 x (u = lse over rows of C, v = lse over rows of CT) with ONE
//            unified wave-per-row kernel: b128 loads, exact max-first LSE,
//            1 exp/element, 32768 waves of parallelism each direction
//   Phase 3: per-batch top-3 of (u_n + v_m - C_nm), threefry pick, emit C[idx]
// ---------------------------------------------------------------------------

#define BATCH 64
#define NPTS  512          // N == M
#define DDIM  256
#define EPS_F       0.1f
#define INV_EPS_F   10.0f
#define NEG_BIG    -1.0e30f

typedef __attribute__((ext_vector_type(16))) __bf16 v16bf;
typedef __attribute__((ext_vector_type(8)))  __bf16 v8bf;
typedef __attribute__((ext_vector_type(4)))  __bf16 v4bf;
typedef __attribute__((ext_vector_type(8)))  float  v8f;

// ---------------- Phase 0: one-time f32 -> bf16 hi/lo split ----------------
__global__ __launch_bounds__(256)
void split_bf16(const float* __restrict__ src, __bf16* __restrict__ hi,
                __bf16* __restrict__ lo) {
  const size_t i = ((size_t)blockIdx.x * 256 + threadIdx.x) * 4;
  const float4 f = *(const float4*)(src + i);
  float ff[4] = {f.x, f.y, f.z, f.w};
  v4bf h, l;
#pragma unroll
  for (int j = 0; j < 4; ++j) {
    __bf16 hv = (__bf16)ff[j];
    h[j] = hv;
    l[j] = (__bf16)(ff[j] - (float)hv);
  }
  *(v4bf*)(hi + i) = h;
  *(v4bf*)(lo + i) = l;
}

// ---------------- Phase 1: batched GEMM, C = 1 - x y^T ----------------
// grid = (M/32, N/64, B), block = 256 (8 waves); wave -> 16x16 C tile
__global__ __launch_bounds__(256)
void sinkhorn_gemm(const __bf16* __restrict__ xh, const __bf16* __restrict__ xl,
                   const __bf16* __restrict__ yh, const __bf16* __restrict__ yl,
                   float* __restrict__ C) {
  const int lane = threadIdx.x & 31;
  const int wid  = threadIdx.x >> 5;
  const int b    = blockIdx.z;
  const int n0   = blockIdx.y * 64 + (wid >> 1) * 16;
  const int m0   = blockIdx.x * 32 + (wid & 1) * 16;

  const int l15  = lane & 15;
  const int hgrp = lane >> 4;                 // lane half
  const int akb  = hgrp * 8;                  // A: K blocks {akb..+7, akb+16..+23}
  const int bkb  = hgrp * 8;                  // B: K blocks {bkb..+7, bkb+16..+23}

  const size_t arow = (size_t)(b * NPTS + n0 + l15) * DDIM;
  const size_t brow = (size_t)(b * NPTS + m0 + l15) * DDIM;

  v8f acc = {};
#pragma unroll
  for (int kc = 0; kc < DDIM; kc += 32) {
    v8bf a0h = *(const v8bf*)(xh + arow + kc + akb);
    v8bf a1h = *(const v8bf*)(xh + arow + kc + akb + 16);
    v8bf a0l = *(const v8bf*)(xl + arow + kc + akb);
    v8bf a1l = *(const v8bf*)(xl + arow + kc + akb + 16);
    v8bf b0h = *(const v8bf*)(yh + brow + kc + bkb);
    v8bf b1h = *(const v8bf*)(yh + brow + kc + bkb + 16);
    v8bf b0l = *(const v8bf*)(yl + brow + kc + bkb);
    v8bf b1l = *(const v8bf*)(yl + brow + kc + bkb + 16);
    v16bf ah, al, bh, bl;
#pragma unroll
    for (int j = 0; j < 8; ++j) {
      ah[j] = a0h[j]; ah[j + 8] = a1h[j];
      al[j] = a0l[j]; al[j + 8] = a1l[j];
      bh[j] = b0h[j]; bh[j + 8] = b1h[j];
      bl[j] = b0l[j]; bl[j + 8] = b1l[j];
    }
    // bf16x3: hi*hi + hi*lo + lo*hi  (f32 accumulate)
    acc = __builtin_amdgcn_wmma_f32_16x16x32_bf16(false, ah, false, bh,
                                                  (short)0, acc, false, false);
    acc = __builtin_amdgcn_wmma_f32_16x16x32_bf16(false, ah, false, bl,
                                                  (short)0, acc, false, false);
    acc = __builtin_amdgcn_wmma_f32_16x16x32_bf16(false, al, false, bh,
                                                  (short)0, acc, false, false);
  }

  // C/D layout: VGPR r -> row (r + 8*hgrp), col (lane&15)
  float* Cb = C + ((size_t)b << 18);
  const int rbase = n0 + hgrp * 8;
  const int ccol  = m0 + l15;
#pragma unroll
  for (int r = 0; r < 8; ++r)
    Cb[(size_t)(rbase + r) * NPTS + ccol] = 1.0f - acc[r];
}

// ---------------- Phase 1b: CT = transpose(C) per batch (LDS tiled) --------
// grid = (8, 8, B), block 256; 64x64 tile, coalesced loads and stores
__global__ __launch_bounds__(256)
void transpose512(const float* __restrict__ src, float* __restrict__ dst) {
  __shared__ float tile[64][65];              // +1 pad: no LDS bank conflicts
  const int b  = blockIdx.z;
  const int n0 = blockIdx.y * 64;
  const int m0 = blockIdx.x * 64;
  const float* S = src + ((size_t)b << 18);
  float*       D = dst + ((size_t)b << 18);
  const int tx  = threadIdx.x & 63;
  const int ty0 = (threadIdx.x >> 6) * 16;
#pragma unroll
  for (int r = 0; r < 16; ++r)
    tile[ty0 + r][tx] = S[(size_t)(n0 + ty0 + r) * NPTS + m0 + tx];
  __syncthreads();
#pragma unroll
  for (int r = 0; r < 16; ++r)
    D[(size_t)(m0 + ty0 + r) * NPTS + n0 + tx] = tile[tx][ty0 + r];
}

// ---------------- Phase 2: unified row-LSE ----------------
// o[b,r] = eps*(LOG_W - lse_k[(w[b,k] - M[b,r,k])/eps]);  one wave per row,
// b128 loads, exact max-first LSE (1 exp/element, no serial chain)
__global__ __launch_bounds__(256)
void sinkhorn_lse(const float* __restrict__ M, const float* __restrict__ w,
                  float* __restrict__ o) {
  const int lane = threadIdx.x & 31;
  const int row  = blockIdx.x * 8 + (threadIdx.x >> 5);   // 0 .. B*512-1
  const int b    = row >> 9;
  const float* Mr = M + ((size_t)row << 9);
  const float* wb = w + (b << 9);

  float t[16];
#pragma unroll
  for (int i = 0; i < 4; ++i) {
    const int k = lane * 4 + i * 128;
    const float4 c  = *(const float4*)(Mr + k);
    const float4 ww = *(const float4*)(wb + k);
    t[i * 4 + 0] = (ww.x - c.x) * INV_EPS_F;
    t[i * 4 + 1] = (ww.y - c.y) * INV_EPS_F;
    t[i * 4 + 2] = (ww.z - c.z) * INV_EPS_F;
    t[i * 4 + 3] = (ww.w - c.w) * INV_EPS_F;
  }
  float mx = t[0];
#pragma unroll
  for (int i = 1; i < 16; ++i) mx = fmaxf(mx, t[i]);
#pragma unroll
  for (int off = 16; off > 0; off >>= 1)
    mx = fmaxf(mx, __shfl_xor(mx, off, 32));
  float s = 0.f;
#pragma unroll
  for (int i = 0; i < 16; ++i) s += __expf(t[i] - mx);
#pragma unroll
  for (int off = 16; off > 0; off >>= 1)
    s += __shfl_xor(s, off, 32);
  if (lane == 0) {
    const float LOG_W = __logf(1.0f / (float)NPTS + 1e-8f);   // log_mu == log_nu
    o[row] = EPS_F * (LOG_W - (__logf(s) + mx));
  }
}

// ---------------- threefry2x32 (jax PRNG) ----------------
__device__ __forceinline__ unsigned rotl32(unsigned x, int r) {
  return (x << r) | (x >> (32 - r));
}
__device__ void threefry2x32(unsigned k0, unsigned k1, unsigned x0, unsigned x1,
                             unsigned& o0, unsigned& o1) {
  const unsigned ks2 = k0 ^ k1 ^ 0x1BD11BDAu;
  const int R0[4] = {13, 15, 26, 6};
  const int R1[4] = {17, 29, 16, 24};
  x0 += k0; x1 += k1;
#pragma unroll
  for (int i = 0; i < 4; ++i) { x0 += x1; x1 = rotl32(x1, R0[i]); x1 ^= x0; }
  x0 += k1; x1 += ks2 + 1u;
#pragma unroll
  for (int i = 0; i < 4; ++i) { x0 += x1; x1 = rotl32(x1, R1[i]); x1 ^= x0; }
  x0 += ks2; x1 += k0 + 2u;
#pragma unroll
  for (int i = 0; i < 4; ++i) { x0 += x1; x1 = rotl32(x1, R0[i]); x1 ^= x0; }
  x0 += k0; x1 += k1 + 3u;
#pragma unroll
  for (int i = 0; i < 4; ++i) { x0 += x1; x1 = rotl32(x1, R1[i]); x1 ^= x0; }
  x0 += k1; x1 += ks2 + 4u;
#pragma unroll
  for (int i = 0; i < 4; ++i) { x0 += x1; x1 = rotl32(x1, R0[i]); x1 ^= x0; }
  x0 += ks2; x1 += k0 + 5u;
  o0 = x0; o1 = x1;
}

// ---------------- Phase 3: top-3 + select + emit cost ----------------
__global__ __launch_bounds__(256)
void sinkhorn_topk(const float* __restrict__ C, const float* __restrict__ u,
                   const float* __restrict__ v, float* __restrict__ out) {
  const int b = blockIdx.x;
  const float* Cb = C + ((size_t)b << 18);
  const float* ub = u + (b << 9);
  const float* vb = v + (b << 9);

  float t0 = NEG_BIG, t1 = NEG_BIG, t2 = NEG_BIG;
  int   i0 = 0, i1 = 0, i2 = 0;
  for (int i = threadIdx.x; i < NPTS * NPTS; i += 256) {
    const int n = i >> 9, m = i & 511;
    const float s = ub[n] + vb[m] - Cb[i];   // monotone proxy for pi = exp(./eps)
    if (s > t2) {
      if (s > t0)      { t2 = t1; i2 = i1; t1 = t0; i1 = i0; t0 = s; i0 = i; }
      else if (s > t1) { t2 = t1; i2 = i1; t1 = s;  i1 = i; }
      else             { t2 = s;  i2 = i; }
    }
  }
  __shared__ float sv[768];
  __shared__ int   si[768];
  sv[threadIdx.x * 3 + 0] = t0; si[threadIdx.x * 3 + 0] = i0;
  sv[threadIdx.x * 3 + 1] = t1; si[threadIdx.x * 3 + 1] = i1;
  sv[threadIdx.x * 3 + 2] = t2; si[threadIdx.x * 3 + 2] = i2;
  __syncthreads();

  if (threadIdx.x == 0) {
    float g0 = NEG_BIG, g1 = NEG_BIG, g2 = NEG_BIG;
    int   j0 = 0, j1 = 0, j2 = 0;
    for (int k = 0; k < 768; ++k) {
      const float s = sv[k];
      const int   i = si[k];
      if (s > g2 || (s == g2 && i < j2)) {
        if (s > g0 || (s == g0 && i < j0))      { g2=g1; j2=j1; g1=g0; j1=j0; g0=s; j0=i; }
        else if (s > g1 || (s == g1 && i < j1)) { g2=g1; j2=j1; g1=s;  j1=i; }
        else                                    { g2=s;  j2=i; }
      }
    }
    // rand_pairs[b] = randint(fold_in(key(0), 7), (B,), 0, 3)[b]
    unsigned f0, f1;
    threefry2x32(0u, 0u, 0u, 7u, f0, f1);               // fold_in(key(0), 7)
    const unsigned c0 = (b < 32) ? (unsigned)b : (unsigned)(b - 32);
    unsigned w0, w1;
    threefry2x32(f0, f1, c0, c0 + 32u, w0, w1);         // random_bits over iota(64)
    const unsigned bits = (b < 32) ? w0 : w1;
    const unsigned r = bits % 3u;
    const int sel = (r == 0u) ? j0 : ((r == 1u) ? j1 : j2);
    out[b] = Cb[sel];
  }
}

// ---------------- init u,v (ws is poisoned each run) ----------------
__global__ void init_uv(float* __restrict__ uv) {
  uv[blockIdx.x * 256 + threadIdx.x] = 0.0f;
}

// ---------------------------------------------------------------------------
extern "C" void kernel_launch(void* const* d_in, const int* in_sizes, int n_in,
                              void* d_out, int out_size, void* d_ws, size_t ws_size,
                              hipStream_t stream) {
  const float* x = (const float*)d_in[0];   // [B, N, D] f32
  const float* y = (const float*)d_in[1];   // [B, M, D] f32

  // ws layout (all offsets power-of-two-ish -> >=32B alignment everywhere)
  const size_t csz = (size_t)BATCH * NPTS * NPTS;        // 16,777,216
  float* C  = (float*)d_ws;                              // 64 MiB
  float* CT = C + csz;                                   // 64 MiB (transposed C)
  float* u  = CT + csz;                                  // 32768 f32
  float* v  = u + BATCH * NPTS;                          // 32768 f32
  __bf16* xh = (__bf16*)(v + BATCH * NPTS);              // 4 x 8.4M bf16 = 64 MiB
  const size_t elems = (size_t)BATCH * NPTS * DDIM;      // 8,388,608
  __bf16* xl = xh + elems;
  __bf16* yh = xl + elems;
  __bf16* yl = yh + elems;
  float* out = (float*)d_out;                            // [B] f32

  // zero u and v (contiguous 2*B*N floats)
  init_uv<<<dim3((2 * BATCH * NPTS) / 256), dim3(256), 0, stream>>>(u);

  // one-time bf16 hi/lo split of x and y
  const int splitBlocks = (int)(elems / (256 * 4));      // 8192
  split_bf16<<<dim3(splitBlocks), dim3(256), 0, stream>>>(x, xh, xl);
  split_bf16<<<dim3(splitBlocks), dim3(256), 0, stream>>>(y, yh, yl);

  // C = 1 - x y^T  (bf16x3 WMMA), then CT = C^T (both stay L2-resident)
  sinkhorn_gemm<<<dim3(NPTS / 32, NPTS / 64, BATCH), dim3(256), 0, stream>>>(
      xh, xl, yh, yl, C);
  transpose512<<<dim3(8, 8, BATCH), dim3(256), 0, stream>>>(C, CT);

  // 50 Sinkhorn iterations; kernel boundaries provide the u->v->u ordering.
  // Both directions use the same coalesced wave-per-row LSE kernel.
  const int lseBlocks = BATCH * NPTS / 8;                // 4096
  for (int it = 0; it < 50; ++it) {
    sinkhorn_lse<<<dim3(lseBlocks), dim3(256), 0, stream>>>(C,  v, u);
    sinkhorn_lse<<<dim3(lseBlocks), dim3(256), 0, stream>>>(CT, u, v);
  }

  sinkhorn_topk<<<dim3(BATCH), dim3(256), 0, stream>>>(C, u, v, out);
}
